// CriticallyFixedProofGNN_10642928959595
// MI455X (gfx1250) — matrix operand order repaired and checked by reading, not compiled, not run on hardware
//
#include <hip/hip_runtime.h>

// ---------------------------------------------------------------------------
// CDNA5 (gfx1250) implementation of the proof-GNN forward pass.
// Dense GEMMs: v_wmma_f32_16x16x32_bf16 with double-buffered async
// global->LDS staging (global_load_async_to_lds_b128 / s_wait_asynccnt).
// ---------------------------------------------------------------------------

typedef __attribute__((ext_vector_type(16))) __bf16 v16bf;
typedef __attribute__((ext_vector_type(8)))  float  v8f;
typedef __attribute__((ext_vector_type(4)))  unsigned int v4u;

union Frag16 { v16bf v; v4u q[2]; };

__device__ __forceinline__ __bf16 f2bf(float f) {
  unsigned u = __float_as_uint(f);
  unsigned r = (u + 0x7FFFu + ((u >> 16) & 1u)) >> 16;   // round-nearest-even
  unsigned short s = (unsigned short)r;
  return __builtin_bit_cast(__bf16, s);
}

__device__ __forceinline__ float warp_sum(float v) {
#pragma unroll
  for (int o = 16; o; o >>= 1) v += __shfl_xor(v, o, 32);
  return v;
}

__device__ __forceinline__ void atomicMaxF(float* addr, float val) {
  int old = __float_as_int(*addr);
  while (__int_as_float(old) < val) {
    int assumed = old;
    old = atomicCAS((int*)addr, assumed, __float_as_int(val));
    if (old == assumed) break;
  }
}

// Async DMA of 16 bytes global -> LDS (ASYNCcnt-tracked, GVS addressing:
// 64-bit SGPR base + 32-bit per-lane VGPR offset; LDS byte address in VGPR).
__device__ __forceinline__ void async_cp_b128(unsigned lds, unsigned goff,
                                              unsigned long long base) {
  asm volatile("global_load_async_to_lds_b128 %0, %1, %2"
               :: "v"(lds), "v"(goff), "s"(base) : "memory");
}

// ---------------------------------------------------------------------------
// WMMA GEMM: C[M,N] = act(A[M,K] @ Bt[N,K]^T + bias)
// A bf16 [M,K] row-major; Bt bf16 [N,K] (contraction contiguous per out col).
// K % 64 == 0, M % 64 == 0, N % 32 == 0.
// Block = 256 threads = 8 waves (4m x 2n), block tile 64x32, K staged by 64.
// Stage s+1 is DMA'd into the alternate LDS buffer while stage s computes.
// ---------------------------------------------------------------------------
__global__ __launch_bounds__(256)
void gemm_bf16_wmma(const __bf16* __restrict__ A, const __bf16* __restrict__ Bt,
                    const float* __restrict__ bias, float* __restrict__ C,
                    __bf16* __restrict__ Cbf, int M, int Nn, int Kk, int act) {
  __shared__ __attribute__((aligned(16))) unsigned char smem[24576]; // 2*(8K+4K)
  const int tid  = threadIdx.x;
  const int lane = tid & 31;
  const int wv   = tid >> 5;
  const int wm   = wv >> 1, wn = wv & 1;
  const int mBase = blockIdx.x * 64;
  const int nBase = blockIdx.y * 32;
  const int l15 = lane & 15;
  const int hi  = lane >> 4;

  typedef __attribute__((address_space(3))) unsigned char lds_t;
  const unsigned ldsBase = (unsigned)(unsigned long long)(lds_t*)&smem[0];
  const unsigned long long aBase = (unsigned long long)A;
  const unsigned long long bBase = (unsigned long long)Bt;

  // One stage = A tile 64 rows x 64 cols (8KB) + B tile 32 rows x 64 (4KB).
  // Per thread: 2 x b128 for A (one row, 32B) + 1 x b128 for B.
  auto issue = [&](int k0, int bsel) {
    const unsigned bo = ldsBase + (bsel ? 12288u : 0u);
    {
      const int row = tid >> 2, cc = (tid & 3) * 2;          // A: 64 rows, 8 chunks/row
      unsigned lds0 = bo + row * 128u + cc * 16u;
      unsigned go = (unsigned)((mBase + row) * Kk + k0) * 2u + cc * 16u;
      async_cp_b128(lds0,      go,      aBase);
      async_cp_b128(lds0 + 16, go + 16, aBase);
    }
    {
      const int row = tid >> 3, cc = tid & 7;                // B: 32 rows, 8 chunks/row
      unsigned lds0 = bo + 8192u + row * 128u + cc * 16u;
      unsigned go = (unsigned)((nBase + row) * Kk + k0) * 2u + cc * 16u;
      async_cp_b128(lds0, go, bBase);
    }
  };

  v8f acc = {0.f, 0.f, 0.f, 0.f, 0.f, 0.f, 0.f, 0.f};
  const int S = Kk >> 6;
  issue(0, 0);
  for (int s = 0; s < S; ++s) {
    if (s + 1 < S) {
      issue((s + 1) << 6, (s + 1) & 1);
      asm volatile("s_wait_asynccnt 0x3" ::: "memory");  // stage s landed, s+1 in flight
    } else {
      asm volatile("s_wait_asynccnt 0x0" ::: "memory");
    }
    __syncthreads();                                     // stage s visible to all waves
    const unsigned char* bufA = smem + ((s & 1) ? 12288 : 0);
    const unsigned char* bufB = bufA + 8192;
#pragma unroll
    for (int kb = 0; kb < 64; kb += 32) {
      Frag16 a, b;
      // A fragment (ISA 7.12.2 16-bit A 16x32): row wm*16+l15,
      // K pairs {base..base+7, base+16..base+23}, base = hi?8:0.
      const unsigned char* ap = bufA + (wm * 16 + l15) * 128 + (kb + (hi ? 8 : 0)) * 2;
      a.q[0] = *(const v4u*)ap;
      a.q[1] = *(const v4u*)(ap + 32);
      // B fragment: out col wn*16+l15, K {base..base+7, base+16..base+23}, base = hi?8:0.
      const unsigned char* bp = bufB + (wn * 16 + l15) * 128 + (kb + (hi ? 8 : 0)) * 2;
      b.q[0] = *(const v4u*)bp;
      b.q[1] = *(const v4u*)(bp + 32);
      acc = __builtin_amdgcn_wmma_f32_16x16x32_bf16(false, a.v, false, b.v,
                                                    (short)0, acc, false, false);
    }
    __syncthreads();                                     // done reading buf before reuse
  }

  // C layout: VGPR r -> row m0 + r + 8*hi, col n0 + l15
  const int m0 = mBase + wm * 16;
  const int col = nBase + wn * 16 + l15;
  const float bv = bias ? bias[col] : 0.f;
#pragma unroll
  for (int r = 0; r < 8; ++r) {
    float v = acc[r] + bv;
    if (act == 1) v = fmaxf(v, 0.f);
    long o = (long)(m0 + r + hi * 8) * Nn + col;
    C[o] = v;
    if (Cbf) Cbf[o] = f2bf(v);
  }
}

// --------------------------- small utility kernels -------------------------
__global__ void fill_kernel(float* p, float v, long n) {
  long i = (long)blockIdx.x * 256 + threadIdx.x;
  if (i < n) p[i] = v;
}
__global__ void cast_kernel(const float* x, __bf16* y, long n) {
  long i = (long)blockIdx.x * 256 + threadIdx.x;
  if (i < n) y[i] = f2bf(x[i]);
}
// W [K,N] f32 -> Wt [N,K] bf16 (coalesced writes)
__global__ void tcast_kernel(const float* W, __bf16* Wt, int K, int Nn) {
  int idx = blockIdx.x * 256 + threadIdx.x;
  if (idx >= K * Nn) return;
  int nn = idx / K, kk = idx - nn * K;
  Wt[idx] = f2bf(W[(long)kk * Nn + nn]);
}

// LayerNorm (optionally + residual R), one block per row, blockDim == W
__global__ void ln_kernel(const float* __restrict__ X, const float* __restrict__ R,
                          const float* __restrict__ g, const float* __restrict__ b,
                          float* __restrict__ Y, __bf16* __restrict__ Ybf, int W) {
  __shared__ float red[512];
  int n = blockIdx.x, t = threadIdx.x;
  float x = X[(long)n * W + t] + (R ? R[(long)n * W + t] : 0.f);
  red[t] = x; __syncthreads();
  for (int s = W >> 1; s; s >>= 1) { if (t < s) red[t] += red[t + s]; __syncthreads(); }
  float mean = red[0] / W; __syncthreads();
  float d = x - mean;
  red[t] = d * d; __syncthreads();
  for (int s = W >> 1; s; s >>= 1) { if (t < s) red[t] += red[t + s]; __syncthreads(); }
  float var = red[0] / W;
  float y = d * rsqrtf(var + 1e-5f) * g[t] + b[t];
  Y[(long)n * W + t] = y;
  if (Ybf) Ybf[(long)n * W + t] = f2bf(y);
}

// --------------------------- spectral pathway ------------------------------
__global__ void spec_filter_kernel(const float* eigvals, const unsigned char* mask,
                                   const float* w1, const float* b1,
                                   const float* w2, const float* b2, float* f) {
  __shared__ float t1[8];
  int t = threadIdx.x;
  if (t < 8) {
    float a = b1[t];
    for (int k = 0; k < 16; ++k) a += eigvals[k] * w1[k * 8 + t];
    t1[t] = fmaxf(a, 0.f);
  }
  __syncthreads();
  if (t < 16) {
    float a = b2[t];
    for (int c = 0; c < 8; ++c) a += t1[c] * w2[c * 16 + t];
    a = tanhf(a);
    f[t] = mask[t] ? a : 0.f;
  }
}
__global__ __launch_bounds__(256)
void xfreq_kernel(const float* ev, const float* x, float* xfreq) {
  __shared__ float red[256];
  int kc = blockIdx.x, kk = kc / 22, c = kc - kk * 22, t = threadIdx.x;
  float acc = 0.f;
  for (int n = t; n < 4096; n += 256) acc += ev[(long)n * 16 + kk] * x[(long)n * 22 + c];
  red[t] = acc; __syncthreads();
  for (int s = 128; s; s >>= 1) { if (t < s) red[t] += red[t + s]; __syncthreads(); }
  if (t == 0) xfreq[kc] = red[0];
}
__global__ void m16_kernel(const float* f, const float* xfreq, const float* pw, float* M16) {
  int kk = blockIdx.x, j = threadIdx.x;
  float acc = 0.f;
  for (int c = 0; c < 22; ++c) acc += xfreq[kk * 22 + c] * pw[(long)c * 256 + j];
  M16[kk * 256 + j] = acc * f[kk];
}
__global__ void hspec_kernel(const float* ev, const float* M16, const float* bias,
                             float* h, __bf16* hbf) {
  __shared__ float e[16];
  int n = blockIdx.x, j = threadIdx.x;
  if (j < 16) e[j] = ev[(long)n * 16 + j];
  __syncthreads();
  float acc = bias[j];
#pragma unroll
  for (int k = 0; k < 16; ++k) acc += e[k] * M16[k * 256 + j];
  h[(long)n * 256 + j] = acc; hbf[(long)n * 256 + j] = f2bf(acc);
}

// --------------------------- spatial (GAT) ---------------------------------
__global__ void inproj_kernel(const float* x, const float* w, const float* b,
                              float* h, __bf16* hbf) {
  __shared__ float xr_[22];
  int n = blockIdx.x, j = threadIdx.x;
  if (j < 22) xr_[j] = x[(long)n * 22 + j];
  __syncthreads();
  float acc = b[j];
#pragma unroll
  for (int c = 0; c < 22; ++c) acc += xr_[c] * w[(long)c * 256 + j];
  h[(long)n * 256 + j] = acc; hbf[(long)n * 256 + j] = f2bf(acc);
}
// ee3[t][j] = edge_emb[t] @ We + be  (only 3 distinct edge types!)
__global__ void ee3_kernel(const float* emb, const float* We, const float* be, float* ee3) {
  int idx = blockIdx.x * 256 + threadIdx.x;
  if (idx >= 3072) return;
  int t = idx >> 10, j = idx & 1023;
  float acc = be[j];
#pragma unroll
  for (int c = 0; c < 8; ++c) acc += emb[t * 8 + c] * We[(long)c * 1024 + j];
  ee3[idx] = acc;
}
__global__ __launch_bounds__(128)
void edge_score_kernel(const int* src_, const int* dst_, const int* eattr,
                       const float* xl, const float* xr, const float* ee3,
                       const float* att, float* aE, float* amax) {
  int e = blockIdx.x, h = threadIdx.x >> 5, lane = threadIdx.x & 31;
  int s = src_[e], d2 = dst_[e], t = eattr[e];
  const float* pl = xl + (long)s * 1024 + h * 256;
  const float* pr = xr + (long)d2 * 1024 + h * 256;
  const float* pe = ee3 + (long)t * 1024 + h * 256;
  const float* pa = att + h * 256;
  float acc = 0.f;
#pragma unroll
  for (int tt = 0; tt < 8; ++tt) {
    int d = tt * 32 + lane;
    float m = pl[d] + pr[d] + pe[d];
    m = m > 0.f ? m : 0.2f * m;          // leaky_relu(0.2)
    acc += m * pa[d];
  }
  acc = warp_sum(acc);
  if (lane == 0) { aE[(long)e * 4 + h] = acc; atomicMaxF(&amax[d2 * 4 + h], acc); }
}
__global__ __launch_bounds__(128)
void edge_accum_kernel(const int* src_, const int* dst_, const float* aE,
                       const float* amax, const float* xl, float* den, float* outacc) {
  int e = blockIdx.x, h = threadIdx.x >> 5, lane = threadIdx.x & 31;
  int s = src_[e], d2 = dst_[e];
  float ex = __expf(aE[(long)e * 4 + h] - amax[d2 * 4 + h]);
  if (lane == 0) atomicAdd(&den[d2 * 4 + h], ex);
  const float* pl = xl + (long)s * 1024 + h * 256;
  float* po = outacc + (long)d2 * 1024 + h * 256;
#pragma unroll
  for (int tt = 0; tt < 8; ++tt) { int d = tt * 32 + lane; atomicAdd(&po[d], ex * pl[d]); }
}
__global__ void gat_finalize_ln(const float* outacc, const float* den, const float* bias,
                                const float* hin, const float* g, const float* b,
                                float* hout, __bf16* hout_bf) {
  __shared__ float red[256];
  int n = blockIdx.x, d = threadIdx.x;
  float v = 0.f;
#pragma unroll
  for (int h = 0; h < 4; ++h)
    v += outacc[(long)n * 1024 + h * 256 + d] / (den[n * 4 + h] + 1e-16f);
  float x = v * 0.25f + bias[d] + hin[(long)n * 256 + d];
  red[d] = x; __syncthreads();
  for (int s = 128; s; s >>= 1) { if (d < s) red[d] += red[d + s]; __syncthreads(); }
  float mean = red[0] / 256.f; __syncthreads();
  float dd = x - mean;
  red[d] = dd * dd; __syncthreads();
  for (int s = 128; s; s >>= 1) { if (d < s) red[d] += red[d + s]; __syncthreads(); }
  float var = red[0] / 256.f;
  float y = dd * rsqrtf(var + 1e-5f) * g[d] + b[d];
  hout[(long)n * 256 + d] = y; hout_bf[(long)n * 256 + d] = f2bf(y);
}

// --------------------------- temporal attention ----------------------------
// steps are sorted: valid keys for query i are j with step[j] in [s_i-5, s_i];
// one wave per (query, head) runs an online-softmax flash loop over that range.
__global__ __launch_bounds__(256)
void attn_kernel(const float* q, const float* k, const float* v,
                 const unsigned char* derived, const int* steps, float* o) {
  int gw = blockIdx.x * 8 + (threadIdx.x >> 5);
  int lane = threadIdx.x & 31;
  int i = gw >> 2, h = gw & 3;
  int si = steps[i];
  int lo = 0, hi_ = 4096;
  while (lo < hi_) { int mid = (lo + hi_) >> 1; if (steps[mid] < si - 5) lo = mid + 1; else hi_ = mid; }
  int jlo = lo;
  lo = 0; hi_ = 4096;
  while (lo < hi_) { int mid = (lo + hi_) >> 1; if (steps[mid] <= si) lo = mid + 1; else hi_ = mid; }
  int jhi = lo - 1;
  const float* qp = q + (long)i * 256 + h * 64 + lane * 2;
  float q0 = qp[0], q1 = qp[1];
  float mrun = -1e30f, lrun = 0.f, o0 = 0.f, o1 = 0.f;
  for (int j = jlo; j <= jhi; ++j) {
    if (!(derived[j] || j == i)) continue;
    const float* kp = k + (long)j * 256 + h * 64 + lane * 2;
    float p = q0 * kp[0] + q1 * kp[1];
    p = warp_sum(p) * 0.125f;                       // 1/sqrt(64)
    float mn = fmaxf(mrun, p);
    float sc = __expf(mrun - mn);
    float w = __expf(p - mn);
    const float* vp = v + (long)j * 256 + h * 64 + lane * 2;
    lrun = lrun * sc + w;
    o0 = o0 * sc + w * vp[0];
    o1 = o1 * sc + w * vp[1];
    mrun = mn;
  }
  float inv = lrun > 0.f ? 1.f / lrun : 0.f;
  float* op = o + (long)i * 256 + h * 64 + lane * 2;
  op[0] = o0 * inv; op[1] = o1 * inv;
}

// --------------------------- fusion / heads --------------------------------
__global__ void gated_kernel(const float* gpre, const float* path, float* fi, int gi) {
  int n = blockIdx.x, d = threadIdx.x;
  float g = 1.f / (1.f + __expf(-gpre[(long)n * 256 + d]));
  fi[(long)n * 768 + gi * 256 + d] = g * path[(long)n * 256 + d];
}
__global__ void addhf_kernel(const float* fo, const float* hspat, float* hf, __bf16* hf_bf) {
  int n = blockIdx.x, d = threadIdx.x;
  float v = fo[(long)n * 256 + d] + hspat[(long)n * 256 + d];
  hf[(long)n * 256 + d] = v; hf_bf[(long)n * 256 + d] = f2bf(v);
}
__global__ __launch_bounds__(256)
void score_kernel(const float* s1n, const float* w, const float* b, float* out) {
  int n = blockIdx.x * 8 + (threadIdx.x >> 5), lane = threadIdx.x & 31;
  float acc = 0.f;
#pragma unroll
  for (int t = 0; t < 8; ++t) { int d = t * 32 + lane; acc += s1n[(long)n * 256 + d] * w[d]; }
  acc = warp_sum(acc);
  if (lane == 0) out[n] = acc + b[0];
}
__global__ void seg_kernel(const float* hf, const int* batch, float* sums, float* cnt) {
  int n = blockIdx.x, d = threadIdx.x;
  int g = batch[n];
  atomicAdd(&sums[(long)g * 256 + d], hf[(long)n * 256 + d]);
  if (d == 0) atomicAdd(&cnt[g], 1.f);
}
__global__ void value_kernel(const float* sums, const float* cnt, const int* batch,
                             const float* w1, const float* b1,
                             const float* w2, const float* b2, float* out) {
  __shared__ float ge[256];
  __shared__ float v1[128];
  int t = threadIdx.x;
  int b0 = batch[0];
  float c = fmaxf(cnt[b0], 1.f);
  for (int d = t; d < 256; d += 128) ge[d] = sums[(long)b0 * 256 + d] / c;
  __syncthreads();
  float acc = b1[t];
  for (int cc = 0; cc < 256; ++cc) acc += ge[cc] * w1[(long)cc * 128 + t];
  v1[t] = fmaxf(acc, 0.f);
  __syncthreads();
  if (t == 0) {
    float s = b2[0];
    for (int j = 0; j < 128; ++j) s += v1[j] * w2[j];
    out[0] = 1.f / (1.f + __expf(-s));
  }
}

// ---------------------------------------------------------------------------
// Host orchestration
// ---------------------------------------------------------------------------
enum {
  IN_X = 0, IN_EIDX, IN_EATTR, IN_EIGV, IN_EIGVAL, IN_EMASK, IN_DMASK, IN_STEPS, IN_BATCH,
  P_FUS_CTX_L1_B, P_FUS_CTX_L1_W, P_FUS_CTX_L2_B, P_FUS_CTX_L2_W, P_FUS_CTX_LN_B, P_FUS_CTX_LN_G,
  P_G_BASE,                                      // gates: 3 x {l1.b,l1.w,l2.b,l2.w,ln1.b,ln1.g}
  P_MLP_L1_B = P_G_BASE + 18, P_MLP_L1_W, P_MLP_L2_B, P_MLP_L2_W,
  P_MLP_LN1_B, P_MLP_LN1_G, P_MLP_LN2_B, P_MLP_LN2_G,
  P_SC_L1_B, P_SC_L1_W, P_SC_L2_B, P_SC_L2_W, P_SC_LN_B, P_SC_LN_G,
  P_EDGE_EMB, P_INPROJ_B, P_INPROJ_W,
  P_LYR_BASE,                                    // layers: 3 x {We,Wl,Wr,att,be,bias,bl,br}
  P_NRM_BASE = P_LYR_BASE + 24,                  // norms: 3 x {b,g}
  P_FG1_B = P_NRM_BASE + 6, P_FG1_W, P_FG2_B, P_FG2_W, P_PROJ_B, P_PROJ_W,
  P_T_K_B, P_T_K_W, P_T_LN_B, P_T_LN_G, P_T_O_B, P_T_O_W, P_T_Q_B, P_T_Q_W, P_T_V_B, P_T_V_W,
  P_V_L1_B, P_V_L1_W, P_V_L2_B, P_V_L2_W
};

extern "C" void kernel_launch(void* const* d_in, const int* in_sizes, int n_in,
                              void* d_out, int out_size, void* d_ws, size_t ws_size,
                              hipStream_t stream) {
  (void)in_sizes; (void)n_in; (void)out_size; (void)ws_size;
  constexpr int Nn = 4096, Dd = 256, Ee = 131072, Gg = 8;

  auto F  = [&](int i) { return (const float*)d_in[i]; };
  auto I  = [&](int i) { return (const int*)d_in[i]; };
  auto Bc = [&](int i) { return (const unsigned char*)d_in[i]; };

  char* wsp = (char*)d_ws; size_t off = 0;
  auto alloc = [&](size_t bytes) { void* p = wsp + off; off = (off + bytes + 255) & ~size_t(255); return p; };

  // f32 scratch
  float* fbuf   = (float*)alloc(16 * 4);
  float* xfreq  = (float*)alloc(16 * 22 * 4);
  float* M16    = (float*)alloc(16 * 256 * 4);
  float* hspec  = (float*)alloc((long)Nn * Dd * 4);
  float* hA     = (float*)alloc((long)Nn * Dd * 4);
  float* hB     = (float*)alloc((long)Nn * Dd * 4);
  float* htemp  = (float*)alloc((long)Nn * Dd * 4);
  float* qb     = (float*)alloc((long)Nn * Dd * 4);
  float* kb     = (float*)alloc((long)Nn * Dd * 4);
  float* vb     = (float*)alloc((long)Nn * Dd * 4);
  float* obuf   = (float*)alloc((long)Nn * Dd * 4);
  float* s256a  = (float*)alloc((long)Nn * Dd * 4);
  float* s256b  = (float*)alloc((long)Nn * Dd * 4);
  float* s512   = (float*)alloc((long)Nn * 512 * 4);
  float* fi     = (float*)alloc((long)Nn * 768 * 4);
  float* u64b   = (float*)alloc((long)Nn * 64 * 4);
  float* xl     = (float*)alloc((long)Nn * 1024 * 4);
  float* xr     = (float*)alloc((long)Nn * 1024 * 4);
  float* outacc = (float*)alloc((long)Nn * 1024 * 4);
  float* ee3    = (float*)alloc(3 * 1024 * 4);
  float* aE     = (float*)alloc((long)Ee * 4 * 4);
  float* amax   = (float*)alloc((long)Nn * 4 * 4);
  float* den    = (float*)alloc((long)Nn * 4 * 4);
  float* sums   = (float*)alloc(Gg * Dd * 4);
  float* cntb   = (float*)alloc(Gg * 4);
  // bf16 scratch
  __bf16* hspec_bf = (__bf16*)alloc((long)Nn * Dd * 2);
  __bf16* hA_bf    = (__bf16*)alloc((long)Nn * Dd * 2);
  __bf16* hB_bf    = (__bf16*)alloc((long)Nn * Dd * 2);
  __bf16* htemp_bf = (__bf16*)alloc((long)Nn * Dd * 2);
  __bf16* obf      = (__bf16*)alloc((long)Nn * Dd * 2);
  __bf16* un_bf    = (__bf16*)alloc((long)Nn * 64 * 2);
  __bf16* t1n_bf   = (__bf16*)alloc((long)Nn * 512 * 2);
  __bf16* fi_bf    = (__bf16*)alloc((long)Nn * 768 * 2);
  __bf16* hf_bf    = (__bf16*)alloc((long)Nn * Dd * 2);
  __bf16* wT       = (__bf16*)alloc(768 * 512 * 2);     // largest weight

  float* out_scores = (float*)d_out;
  float* out_hf     = out_scores + Nn;
  float* out_val    = out_hf + (long)Nn * Dd;

  const int* src = I(IN_EIDX);
  const int* dst = I(IN_EIDX) + Ee;

  auto gemm = [&](const __bf16* A, const __bf16* Bt, const float* bias,
                  float* C, __bf16* Cbf, int M, int Nc, int Kc, int act) {
    dim3 g(M / 64, Nc / 32);
    gemm_bf16_wmma<<<g, 256, 0, stream>>>(A, Bt, bias, C, Cbf, M, Nc, Kc, act);
  };
  auto tcast = [&](const float* W, int Kc, int Nc) {
    int n = Kc * Nc;
    tcast_kernel<<<(n + 255) / 256, 256, 0, stream>>>(W, wT, Kc, Nc);
  };
  auto fill = [&](float* p, float v, long n) {
    fill_kernel<<<(int)((n + 255) / 256), 256, 0, stream>>>(p, v, n);
  };

  // ---- spectral ----
  spec_filter_kernel<<<1, 32, 0, stream>>>(F(IN_EIGVAL), Bc(IN_EMASK),
      F(P_FG1_W), F(P_FG1_B), F(P_FG2_W), F(P_FG2_B), fbuf);
  xfreq_kernel<<<16 * 22, 256, 0, stream>>>(F(IN_EIGV), F(IN_X), xfreq);
  m16_kernel<<<16, 256, 0, stream>>>(fbuf, xfreq, F(P_PROJ_W), M16);
  hspec_kernel<<<Nn, 256, 0, stream>>>(F(IN_EIGV), M16, F(P_PROJ_B), hspec, hspec_bf);

  // ---- spatial: input projection then 3 GAT layers ----
  inproj_kernel<<<Nn, 256, 0, stream>>>(F(IN_X), F(P_INPROJ_W), F(P_INPROJ_B), hA, hA_bf);
  float* hcur = hA;       __bf16* hcur_bf = hA_bf;
  float* hnxt = hB;       __bf16* hnxt_bf = hB_bf;
  for (int l = 0; l < 3; ++l) {
    int L = P_LYR_BASE + l * 8;   // +0 We, +1 Wl, +2 Wr, +3 att, +4 be, +5 bias, +6 bl, +7 br
    ee3_kernel<<<12, 256, 0, stream>>>(F(P_EDGE_EMB), F(L + 0), F(L + 4), ee3);
    tcast(F(L + 1), 256, 1024);
    gemm(hcur_bf, wT, F(L + 6), xl, nullptr, Nn, 1024, 256, 0);
    tcast(F(L + 2), 256, 1024);
    gemm(hcur_bf, wT, F(L + 7), xr, nullptr, Nn, 1024, 256, 0);
    fill(amax, -1e30f, (long)Nn * 4);
    fill(den, 0.f, (long)Nn * 4);
    fill(outacc, 0.f, (long)Nn * 1024);
    edge_score_kernel<<<Ee, 128, 0, stream>>>(src, dst, I(IN_EATTR), xl, xr, ee3,
                                              F(L + 3), aE, amax);
    edge_accum_kernel<<<Ee, 128, 0, stream>>>(src, dst, aE, amax, xl, den, outacc);
    int NB = P_NRM_BASE + l * 2;  // +0 b, +1 g
    gat_finalize_ln<<<Nn, 256, 0, stream>>>(outacc, den, F(L + 5), hcur,
                                            F(NB + 1), F(NB + 0), hnxt, hnxt_bf);
    float* tf = hcur; hcur = hnxt; hnxt = tf;
    __bf16* tb = hcur_bf; hcur_bf = hnxt_bf; hnxt_bf = tb;
  }
  float* hspat = hcur; __bf16* hspat_bf = hcur_bf;

  // ---- temporal ----
  tcast(F(P_T_Q_W), 256, 256); gemm(hspec_bf, wT, F(P_T_Q_B), qb, nullptr, Nn, 256, 256, 0);
  tcast(F(P_T_K_W), 256, 256); gemm(hspec_bf, wT, F(P_T_K_B), kb, nullptr, Nn, 256, 256, 0);
  tcast(F(P_T_V_W), 256, 256); gemm(hspec_bf, wT, F(P_T_V_B), vb, nullptr, Nn, 256, 256, 0);
  attn_kernel<<<Nn * 4 / 8, 256, 0, stream>>>(qb, kb, vb, Bc(IN_DMASK), I(IN_STEPS), obuf);
  cast_kernel<<<(Nn * Dd + 255) / 256, 256, 0, stream>>>(obuf, obf, (long)Nn * Dd);
  tcast(F(P_T_O_W), 256, 256); gemm(obf, wT, F(P_T_O_B), s256a, nullptr, Nn, 256, 256, 0);
  ln_kernel<<<Nn, 256, 0, stream>>>(s256a, hspec, F(P_T_LN_G), F(P_T_LN_B),
                                    htemp, htemp_bf, 256);

  // ---- fusion: 3 gates + MLP ----
  const float*  pf[3]  = { hspec, hspat, htemp };
  const __bf16* pbf[3] = { hspec_bf, hspat_bf, htemp_bf };
  for (int gi = 0; gi < 3; ++gi) {
    int GB = P_G_BASE + gi * 6;   // +0 l1.b, +1 l1.w, +2 l2.b, +3 l2.w, +4 ln1.b, +5 ln1.g
    tcast(F(GB + 1), 256, 64);
    gemm(pbf[gi], wT, F(GB + 0), u64b, nullptr, Nn, 64, 256, 1);
    ln_kernel<<<Nn, 64, 0, stream>>>(u64b, nullptr, F(GB + 5), F(GB + 4), u64b, un_bf, 64);
    tcast(F(GB + 3), 64, 256);
    gemm(un_bf, wT, F(GB + 2), s256b, nullptr, Nn, 256, 64, 0);
    gated_kernel<<<Nn, 256, 0, stream>>>(s256b, pf[gi], fi, gi);
  }
  cast_kernel<<<((long)Nn * 768 + 255) / 256, 256, 0, stream>>>(fi, fi_bf, (long)Nn * 768);
  tcast(F(P_MLP_L1_W), 768, 512);
  gemm(fi_bf, wT, F(P_MLP_L1_B), s512, nullptr, Nn, 512, 768, 1);
  ln_kernel<<<Nn, 512, 0, stream>>>(s512, nullptr, F(P_MLP_LN1_G), F(P_MLP_LN1_B),
                                    s512, t1n_bf, 512);
  tcast(F(P_MLP_L2_W), 512, 256);
  gemm(t1n_bf, wT, F(P_MLP_L2_B), s256a, nullptr, Nn, 256, 512, 0);
  ln_kernel<<<Nn, 256, 0, stream>>>(s256a, nullptr, F(P_MLP_LN2_G), F(P_MLP_LN2_B),
                                    s256a, nullptr, 256);
  addhf_kernel<<<Nn, 256, 0, stream>>>(s256a, hspat, out_hf, hf_bf);

  // ---- scorer ----
  tcast(F(P_SC_L1_W), 256, 256);
  gemm(hf_bf, wT, F(P_SC_L1_B), s256b, nullptr, Nn, 256, 256, 1);
  ln_kernel<<<Nn, 256, 0, stream>>>(s256b, nullptr, F(P_SC_LN_G), F(P_SC_LN_B),
                                    s256b, nullptr, 256);
  score_kernel<<<Nn / 8, 256, 0, stream>>>(s256b, F(P_SC_L2_W), F(P_SC_L2_B), out_scores);

  // ---- value head ----
  fill(sums, 0.f, Gg * Dd);
  fill(cntb, 0.f, Gg);
  seg_kernel<<<Nn, 256, 0, stream>>>(out_hf, I(IN_BATCH), sums, cntb);
  value_kernel<<<1, 128, 0, stream>>>(sums, cntb, I(IN_BATCH),
      F(P_V_L1_W), F(P_V_L1_B), F(P_V_L2_W), F(P_V_L2_B), out_val);
}
